// DecouplingGCN_69578470195326
// MI455X (gfx1250) — compile-verified, matrix operand order checked by reading, not run on hardware
//
#include <hip/hip_runtime.h>
#include <hip/hip_bf16.h>

typedef __attribute__((ext_vector_type(16))) __bf16 v16bf;
typedef __attribute__((ext_vector_type(8)))  float  v8f;

#define D_MODEL 256
#define DK      768
#define NB      32
#define TT      256
#define VVV     19
#define MROWS   (NB*TT*VVV)   // 155648
#define NT      (NB*TT)       // 8192

// workspace byte offsets
#define WS_WT      0u          // 768*256 bf16 (W transposed)
#define WS_STATS0  393216u     // 1536 f32: sum[768], sumsq[768]
#define WS_STATS1  399360u     // 512 f32:  sum[256], sumsq[256]
#define WS_AN      401408u     // 24*361 f32 normalized adjacency
#define WS_CS      436064u     // 24*19 f32 column sums of An
#define WS_S       437888u     // 768 f32 BN scale
#define WS_B       440960u     // 768 f32 BN shift
#define WS_S1      444032u     // 256 f32
#define WS_B1      445056u     // 256 f32
#define WS_Y       458752u     // MROWS*768 bf16 intermediate y

// ---------------------------------------------------------------- zero stats
__global__ void k_zero(float* __restrict__ p, int n) {
    int i = blockIdx.x * blockDim.x + threadIdx.x;
    if (i < n) p[i] = 0.f;
}

// ------------------------------------------- W (256x768 f32) -> Wt (768x256 bf16)
__global__ void k_wt(const float* __restrict__ w, __bf16* __restrict__ wt) {
    int i = blockIdx.x * blockDim.x + threadIdx.x;   // 0 .. 768*256-1
    int d = i >> 8, c = i & 255;                     // wt[d][c] = w[c][d]
    wt[i] = (__bf16)w[c * 768 + d];
}

// ---------------------------------------------------------------- main GEMM
// y[m,d] = sum_c x[m,c] * W[c,d] + bias[d], m=(n,t,v). bf16 WMMA, f32 acc.
// Block tile 128x128, K-step 32, 8 waves (wave32): each wave 32Mx64N = 2x4 WMMA tiles.
__global__ __launch_bounds__(256)
void k_gemm(const float* __restrict__ x, const __bf16* __restrict__ wt,
            const float* __restrict__ bias, __bf16* __restrict__ y,
            float* __restrict__ st_sum, float* __restrict__ st_sq) {
    __shared__ __align__(16) __bf16 Xs[128 * 40];   // 128 rows x 32 bf16, pad 8
    __shared__ __align__(16) __bf16 Wls[128 * 40];
    __shared__ float colS[128];
    __shared__ float colQ[128];

    const int tid  = threadIdx.x;
    const int lane = tid & 31;
    const int wave = tid >> 5;
    const int wm   = wave & 3;      // M sub-block (32 rows)
    const int wn   = wave >> 2;     // N sub-block (64 cols)
    const int l16  = lane & 15;
    const int lh   = lane >> 4;     // half-wave: selects K-chunk per WMMA layout

    const long m0 = (long)blockIdx.x * 128;
    const int  n0 = blockIdx.y * 128;

    v8f zero = {};
    v8f acc[2][4];
    for (int mi = 0; mi < 2; ++mi)
        for (int ni = 0; ni < 4; ++ni) acc[mi][ni] = zero;

    union Pack4 { __bf16 h[4]; uint2 u; };
    union Frag  { uint4 u[2]; v16bf v; };

    for (int kk = 0; kk < 8; ++kk) {
        __syncthreads();
        // stage X tile (128x32 f32 -> bf16): 1024 float4 slots
        for (int j = 0; j < 4; ++j) {
            int idx = tid + 256 * j;
            int row = idx >> 3, c4 = idx & 7;
            const float4 f = ((const float4*)x)[(m0 + row) * 64 + kk * 8 + c4];
            Pack4 p;
            p.h[0] = (__bf16)f.x; p.h[1] = (__bf16)f.y;
            p.h[2] = (__bf16)f.z; p.h[3] = (__bf16)f.w;
            *(uint2*)&Xs[row * 40 + c4 * 4] = p.u;
        }
        // stage Wt tile (128x32 bf16): 512 uint4 slots
        for (int j = 0; j < 2; ++j) {
            int idx = tid + 256 * j;
            int row = idx >> 2, q = idx & 3;
            const uint4 u = *(const uint4*)(wt + (size_t)(n0 + row) * 256 + kk * 32 + q * 8);
            *(uint4*)&Wls[row * 40 + q * 8] = u;
        }
        __syncthreads();

        Frag a[2], b[4];
        for (int mi = 0; mi < 2; ++mi) {
            const char* base = (const char*)Xs + (wm * 32 + mi * 16 + l16) * 80 + lh * 16;
            a[mi].u[0] = *(const uint4*)base;
            a[mi].u[1] = *(const uint4*)(base + 32);
        }
        for (int ni = 0; ni < 4; ++ni) {
            const char* base = (const char*)Wls + (wn * 64 + ni * 16 + l16) * 80 + lh * 16;
            b[ni].u[0] = *(const uint4*)base;
            b[ni].u[1] = *(const uint4*)(base + 32);
        }
        #pragma unroll
        for (int mi = 0; mi < 2; ++mi)
            #pragma unroll
            for (int ni = 0; ni < 4; ++ni)
                acc[mi][ni] = __builtin_amdgcn_wmma_f32_16x16x32_bf16(
                    false, a[mi].v, false, b[ni].v, (short)0, acc[mi][ni], false, false);
    }

    // epilogue: bias add, bf16 store, per-column sum/sumsq through LDS
    if (tid < 128) { colS[tid] = 0.f; colQ[tid] = 0.f; }
    __syncthreads();

    for (int mi = 0; mi < 2; ++mi) {
        for (int ni = 0; ni < 4; ++ni) {
            const int colb = wn * 64 + ni * 16 + l16;
            const int d    = n0 + colb;
            const float bv = bias[d];
            float ps = 0.f, pq = 0.f;
            #pragma unroll
            for (int r = 0; r < 8; ++r) {
                const long m = m0 + wm * 32 + mi * 16 + lh * 8 + r;
                const float v = acc[mi][ni][r] + bv;
                y[(size_t)m * 768 + d] = (__bf16)v;
                ps += v; pq += v * v;
            }
            atomicAdd(&colS[colb], ps);   // ds_add_f32
            atomicAdd(&colQ[colb], pq);
        }
    }
    __syncthreads();
    if (tid < 128) {
        unsafeAtomicAdd(st_sum + n0 + tid, colS[tid]);   // global_atomic_add_f32
        unsafeAtomicAdd(st_sq  + n0 + tid, colQ[tid]);
    }
}

// -------------------------- adjacency normalize + colsums + BN0 affine fold
__global__ void k_prep(const float* __restrict__ para,
                       const float* __restrict__ st_sum, const float* __restrict__ st_sq,
                       const float* __restrict__ gamma,  const float* __restrict__ beta,
                       float* __restrict__ An, float* __restrict__ Cs,
                       float* __restrict__ Sv, float* __restrict__ Bv) {
    __shared__ float Ash[24 * 361];
    const int tid = threadIdx.x;   // 768 threads
    if (tid < 456) {               // 24 (k,g) matrices x 19 rows, L2 row-normalize
        int kg = tid / 19, v = tid % 19;
        const float* row = para + kg * 361 + v * 19;
        float ss = 0.f;
        for (int w = 0; w < 19; ++w) ss += row[w] * row[w];
        float inv = 1.f / (sqrtf(ss) + 1e-4f);
        for (int w = 0; w < 19; ++w) Ash[kg * 361 + v * 19 + w] = row[w] * inv;
    }
    __syncthreads();
    if (tid < 456) {               // column sums (for the BN-shift term)
        int kg = tid / 19, w = tid % 19;
        float s = 0.f;
        for (int v = 0; v < 19; ++v) s += Ash[kg * 361 + v * 19 + w];
        Cs[kg * 19 + w] = s;
    }
    for (int i = tid; i < 24 * 361; i += blockDim.x) An[i] = Ash[i];
    if (tid < 768) {               // BN(y) == y*s + b with bias already folded into y
        const float cnt = (float)MROWS;
        float mean = st_sum[tid] / cnt;
        float var  = st_sq[tid] / cnt - mean * mean;
        float s    = gamma[tid] * rsqrtf(var + 1e-5f);
        Sv[tid] = s;
        Bv[tid] = beta[tid] - mean * s;
    }
}

// ------------- per-(n,t): z[c,w] = sum_k( s*sum_v y*A + b*colsum ) + x residual
__global__ __launch_bounds__(256)
void k_gcn(const __bf16* __restrict__ y, const float* __restrict__ x,
           const float* __restrict__ An, const float* __restrict__ Cs,
           const float* __restrict__ Sv, const float* __restrict__ Bv,
           float* __restrict__ out, float* __restrict__ st1_sum, float* __restrict__ st1_sq) {
    __shared__ __bf16 ylb[19 * 768];       // 29184 B, contiguous chunk of y
    __shared__ float  Ash[24 * 361];       // 34656 B
    __shared__ float  csh[24 * 19];
    __shared__ float  sl[768], bl[768];
    __shared__ float  xl[19 * 256];        // residual chunk

    const int tid = threadIdx.x;
    const long nt = blockIdx.x;            // (n,t) in 0..8191
    const __bf16* yc = y + nt * (19 * 768);
    const float*  xc = x + nt * (19 * 256);

    for (int i = tid; i < 19 * 768; i += 256) ylb[i] = yc[i];
    for (int i = tid; i < 19 * 256; i += 256) xl[i] = xc[i];
    for (int i = tid; i < 24 * 361; i += 256) Ash[i] = An[i];
    for (int i = tid; i < 24 * 19;  i += 256) csh[i] = Cs[i];
    for (int i = tid; i < 768;      i += 256) { sl[i] = Sv[i]; bl[i] = Bv[i]; }
    __syncthreads();

    const int c = tid;        // one channel per thread
    const int g = c & 7;      // adjacency group (tile semantics: A[k,c]=para[k,c%8])
    float z[19];
    #pragma unroll
    for (int w = 0; w < 19; ++w) z[w] = xl[w * 256 + c];   // residual xc[n,c,t,w]

    for (int k = 0; k < 3; ++k) {
        const float sk = sl[k * 256 + c];
        const float bk = bl[k * 256 + c];
        const float* Ak = Ash + (k * 8 + g) * 361;
        const float* ck = csh + (k * 8 + g) * 19;
        float yv[19];
        #pragma unroll
        for (int v = 0; v < 19; ++v) yv[v] = (float)ylb[v * 768 + k * 256 + c];
        #pragma unroll
        for (int w = 0; w < 19; ++w) {
            float acc = 0.f;
            #pragma unroll
            for (int v = 0; v < 19; ++v) acc += yv[v] * Ak[v * 19 + w];
            z[w] += sk * acc + bk * ck[w];
        }
    }

    float ssum = 0.f, ssq = 0.f;
    float* od = out + nt * 19 * 256;       // (n,t,w,c) layout == final output layout
    #pragma unroll
    for (int w = 0; w < 19; ++w) {
        od[w * 256 + c] = z[w];
        ssum += z[w]; ssq += z[w] * z[w];
    }
    unsafeAtomicAdd(st1_sum + c, ssum);
    unsafeAtomicAdd(st1_sq  + c, ssq);
}

// ------------------------------------------------------- BN1 affine finalize
__global__ void k_bn1(const float* __restrict__ st_sum, const float* __restrict__ st_sq,
                      const float* __restrict__ gamma, const float* __restrict__ beta,
                      float* __restrict__ S1, float* __restrict__ B1) {
    int c = threadIdx.x;
    const float cnt = (float)MROWS;
    float mean = st_sum[c] / cnt;
    float var  = st_sq[c] / cnt - mean * mean;
    float s    = gamma[c] * rsqrtf(var + 1e-5f);
    S1[c] = s;
    B1[c] = beta[c] - mean * s;
}

// --------------------------------------------- in-place affine + ReLU on out
__global__ void k_relu(float* __restrict__ out,
                       const float* __restrict__ S1, const float* __restrict__ B1) {
    const long total4 = (long)MROWS * 256 / 4;
    for (long i = blockIdx.x * (long)blockDim.x + threadIdx.x; i < total4;
         i += (long)gridDim.x * blockDim.x) {
        float4 v = ((const float4*)out)[i];
        int c = (int)((i * 4) & 255);
        v.x = fmaxf(v.x * S1[c]     + B1[c],     0.f);
        v.y = fmaxf(v.y * S1[c + 1] + B1[c + 1], 0.f);
        v.z = fmaxf(v.z * S1[c + 2] + B1[c + 2], 0.f);
        v.w = fmaxf(v.w * S1[c + 3] + B1[c + 3], 0.f);
        ((float4*)out)[i] = v;
    }
}

extern "C" void kernel_launch(void* const* d_in, const int* in_sizes, int n_in,
                              void* d_out, int out_size, void* d_ws, size_t ws_size,
                              hipStream_t stream) {
    const float* x    = (const float*)d_in[0];   // (N,T,V,D)
    const float* para = (const float*)d_in[1];   // (3,8,19,19)
    const float* lw   = (const float*)d_in[2];   // (256,768)
    const float* lb   = (const float*)d_in[3];   // (768)
    const float* bng  = (const float*)d_in[4];
    const float* bnb  = (const float*)d_in[5];
    const float* b1g  = (const float*)d_in[6];
    const float* b1b  = (const float*)d_in[7];
    float* out = (float*)d_out;
    char*  ws  = (char*)d_ws;

    __bf16* wt  = (__bf16*)(ws + WS_WT);
    float* st0s = (float*)(ws + WS_STATS0);
    float* st0q = st0s + 768;
    float* st1s = (float*)(ws + WS_STATS1);
    float* st1q = st1s + 256;
    float* An   = (float*)(ws + WS_AN);
    float* Cs   = (float*)(ws + WS_CS);
    float* Sv   = (float*)(ws + WS_S);
    float* Bv   = (float*)(ws + WS_B);
    float* S1   = (float*)(ws + WS_S1);
    float* B1   = (float*)(ws + WS_B1);
    __bf16* y   = (__bf16*)(ws + WS_Y);

    k_zero<<<8, 256, 0, stream>>>(st0s, 2048);                  // stats0 + stats1 (contiguous)
    k_wt<<<768, 256, 0, stream>>>(lw, wt);
    k_gemm<<<dim3(1216, 6), 256, 0, stream>>>(x, wt, lb, y, st0s, st0q);
    k_prep<<<1, 768, 0, stream>>>(para, st0s, st0q, bng, bnb, An, Cs, Sv, Bv);
    k_gcn<<<8192, 256, 0, stream>>>(y, x, An, Cs, Sv, Bv, out, st1s, st1q);
    k_bn1<<<1, 256, 0, stream>>>(st1s, st1q, b1g, b1b, S1, B1);
    k_relu<<<4096, 256, 0, stream>>>(out, S1, B1);
}
